// SynthesisBlock_1546188227181
// MI455X (gfx1250) — matrix operand order repaired and checked by reading, not compiled, not run on hardware
//
#include <hip/hip_runtime.h>
#include <hip/hip_bf16.h>

typedef __attribute__((ext_vector_type(16))) __bf16 v16bf;
typedef __attribute__((ext_vector_type(8)))  __bf16 v8bf;
typedef __attribute__((ext_vector_type(8)))  float  v8f;
typedef __attribute__((ext_vector_type(8)))  unsigned short u16x8;
typedef __attribute__((ext_vector_type(4)))  unsigned int   u32x4;
typedef __attribute__((ext_vector_type(8)))  int            i32x8;
typedef __attribute__((ext_vector_type(4)))  int            i32x4;

#define B_   16
#define CIN  512
#define COUT 512
#define HH   64
#define WW   64
#define LL   512
#define WSTR 40   // LDS row stride in halves (padded: conflict-free b128 loads)

#if __has_builtin(__builtin_amdgcn_tensor_load_to_lds)
#define HAVE_TDM 1
#endif

__device__ __forceinline__ unsigned short f2bf(float f) {
  unsigned int u = __builtin_bit_cast(unsigned int, f);
  u += 0x7FFFu + ((u >> 16) & 1u);   // round-to-nearest-even
  return (unsigned short)(u >> 16);
}

// ---------------- style = latent @ (affine_w * sqrt(1/L)).T + affine_b ------
__global__ void k_style(const float* __restrict__ latent,
                        const float* __restrict__ affine_w,
                        const float* __restrict__ affine_b,
                        float* __restrict__ style) {
  int idx = blockIdx.x * blockDim.x + threadIdx.x;   // [0, 16*512)
  int b = idx >> 9, c = idx & 511;
  const float* lw = latent + (size_t)b * LL;
  const float* aw = affine_w + (size_t)c * LL;
  float s = 0.f;
  for (int l = 0; l < LL; ++l) s += lw[l] * aw[l];
  style[idx] = s * 0.04419417382415922f + affine_b[c];
}

// ---------------- dnorm[b][cout] = rsqrt(sum((w*style)^2) + 1e-8) -----------
__global__ void k_dnorm(const float* __restrict__ conv_w,
                        const float* __restrict__ style,
                        float* __restrict__ dnorm) {
  int wv = threadIdx.x >> 5, ln = threadIdx.x & 31;
  int gid = blockIdx.x * 8 + wv;                     // [0, 16*512)
  int b = gid >> 9, co = gid & 511;
  const float* wr = conv_w + (size_t)co * (CIN * 9);
  const float* st = style + (size_t)b * CIN;
  float s = 0.f;
  for (int i = ln; i < CIN * 9; i += 32) {
    float t = wr[i] * st[i / 9];
    s += t * t;
  }
  for (int m = 16; m >= 1; m >>= 1) s += __shfl_xor(s, m, 32);
  if (ln == 0) dnorm[gid] = rsqrtf(s + 1e-8f);
}

// ------- wmod[b][cout][kk][cin] (bf16) = conv_w * style * dnorm -------------
__global__ void k_wmod(const float* __restrict__ conv_w,
                       const float* __restrict__ style,
                       const float* __restrict__ dnorm,
                       unsigned short* __restrict__ wmod) {
  int idx = blockIdx.x * blockDim.x + threadIdx.x;   // [0, 16*512*512)
  int b  = idx >> 18;
  int co = (idx >> 9) & 511;
  int ci = idx & 511;
  float s = style[(b << 9) + ci] * dnorm[(b << 9) + co];
  const float* wsrc = conv_w + ((size_t)((co << 9) + ci)) * 9;
  size_t obase = (((size_t)(b << 9) + co) * 9) * 512 + ci;
#pragma unroll
  for (int k = 0; k < 9; ++k)
    wmod[obase + (size_t)k * 512] = f2bf(wsrc[k] * s);
}

// ------- content f32 [b][cin][y][x] -> bf16 channel-last [b][y][x][cin] -----
__global__ void k_cvt(const float* __restrict__ content,
                      unsigned short* __restrict__ cbf) {
  __shared__ float tile[64][129];                    // padded: no bank conflicts
  int cc = blockIdx.x;                               // cin chunk of 128 (0..3)
  int y  = blockIdx.y;
  int b  = blockIdx.z;
  int t  = threadIdx.x;
  const float* src = content + ((size_t)b * CIN + cc * 128) * (HH * WW) + y * WW;
#pragma unroll
  for (int i = 0; i < 32; ++i) {
    int idx = i * 256 + t;
    int c = idx >> 6, x = idx & 63;
    tile[x][c] = src[(size_t)c * (HH * WW) + x];
  }
  __syncthreads();
  unsigned short* dst = cbf + (((size_t)b * HH + y) * WW) * CIN + cc * 128;
#pragma unroll
  for (int j = 0; j < 32; ++j) {
    int idx = j * 256 + t;
    int x = idx >> 7, c = idx & 127;
    dst[(size_t)x * CIN + c] = f2bf(tile[x][c]);
  }
}

// ------- implicit-GEMM 3x3 conv, bf16 WMMA, TDM weight staging --------------
__global__ void __launch_bounds__(256)
k_conv(const unsigned short* __restrict__ cbf,
       const unsigned short* __restrict__ wmod,
       const float* __restrict__ noise,
       const float* __restrict__ bias,
       const float* __restrict__ nwp,
       float* __restrict__ out) {
  // weight rows in memory order: row = co*9 + kk, 32 halves + 8 pad (80 B)
  __shared__ unsigned short w_lds[128 * 9 * WSTR];   // 92160 B
  __shared__ unsigned short in_lds[3 * 66 * WSTR];   // 15840 B

  const int y0    = blockIdx.x;          // output row
  const int coutB = blockIdx.y * 128;    // cout tile
  const int b     = blockIdx.z;
  const int tid   = threadIdx.x;
  const int wv    = tid >> 5;            // 8 waves: M sub-tiles of 16
  const int lane  = tid & 31;
  const int lm    = lane & 15;           // A row / B col within tile
  const int h     = lane >> 4;           // lane half

  v8f acc[4] = {};

  // zero the halo columns (x=-1 -> col 0, x=64 -> col 65); never rewritten
  for (int q = tid; q < 24; q += 256) {
    int ky = q / 8, r = q % 8;
    int col = (r & 4) ? 65 : 0;
    int c16 = r & 3;
    u16x8 z = {};
    *(u16x8*)&in_lds[(ky * 66 + col) * WSTR + c16 * 8] = z;
  }

  const unsigned short* wsrc = wmod + ((size_t)(b * 512 + coutB) * 9) * 512;
  const unsigned short* csrc = cbf + (size_t)b * (HH * WW * CIN);
#ifdef HAVE_TDM
  const unsigned lds_base = (unsigned)(size_t)(void*)w_lds;
#endif

  for (int cb = 0; cb < 16; ++cb) {
    __syncthreads();
    const int cin0 = cb * 32;

#ifdef HAVE_TDM
    // TDM: 2D tile, 1152 rows x 32 halves, row stride 512 halves in memory,
    // LDS padding 4 DWORDs after every 16 DWORDs -> 80 B LDS row stride.
    if (wv == 0) {
      unsigned long long ga = (unsigned long long)(size_t)(wsrc + cin0);
      u32x4 g0 = {
        1u,                                               // count=1 (valid D#)
        lds_base,                                         // lds_addr
        (unsigned)(ga & 0xffffffffu),                     // global_addr[31:0]
        (unsigned)((ga >> 32) & 0x01ffffffu) | (2u << 30) // ga[56:32] | type=2
      };
      i32x8 g1 = {
        (1 << 16) | (1 << 20) | (3 << 22) | (3 << 25),    // 2B, pad_en, int=16dw, amt=4dw
        (32 << 16),                                       // tensor_dim0 = 32
        (1152 << 16),                                     // tensor_dim1 = 1152
        (32 << 16),                                       // tile_dim0 = 32
        1152,                                             // tile_dim1 = 1152
        512,                                              // tensor_dim0_stride
        0, 0
      };
      i32x4 gz4 = {0, 0, 0, 0};
      i32x8 gz8 = {0, 0, 0, 0, 0, 0, 0, 0};
      __builtin_amdgcn_tensor_load_to_lds(g0, g1, gz4, gz4, gz8, 0);
    }
#else
    // fallback: manual staging, same layout (row = co*9+kk)
#pragma unroll
    for (int i = 0; i < 18; ++i) {
      int q = i * 256 + tid;
      int co = q / 36, r = q % 36;
      int kk = r >> 2, c16 = r & 3;
      u16x8 v = *(const u16x8*)&wsrc[((size_t)co * 9 + kk) * 512 + cin0 + c16 * 8];
      *(u16x8*)&w_lds[(co * 9 + kk) * WSTR + c16 * 8] = v;
    }
#endif

    // stage input: 3 rows x 64 cols x 32 cin  (768 x 16B chunks)
#pragma unroll
    for (int i = 0; i < 3; ++i) {
      int q = i * 256 + tid;
      int ky = q >> 8, r = q & 255;
      int x = r >> 2, c16 = r & 3;
      int y = y0 + ky - 1;
      u16x8 v = {};
      if (y >= 0 && y < HH)
        v = *(const u16x8*)&csrc[((size_t)y * WW + x) * CIN + cin0 + c16 * 8];
      *(u16x8*)&in_lds[(ky * 66 + x + 1) * WSTR + c16 * 8] = v;
    }

#ifdef HAVE_TDM
    if (wv == 0) __builtin_amdgcn_s_wait_tensorcnt(0);
#endif
    __syncthreads();

#pragma unroll
    for (int kk = 0; kk < 9; ++kk) {
      const int ky = kk / 3, kx = kk % 3;
      // A fragment: lane h -> K runs [h*8, h*8+7] and [16+h*8, 16+h*8+7]
      const unsigned short* ap = &w_lds[((wv * 16 + lm) * 9 + kk) * WSTR];
      v8bf alo = __builtin_bit_cast(v8bf, *(const u16x8*)&ap[h * 8]);
      v8bf ahi = __builtin_bit_cast(v8bf, *(const u16x8*)&ap[16 + h * 8]);
      v16bf A = __builtin_shufflevector(alo, ahi,
          0,1,2,3,4,5,6,7,8,9,10,11,12,13,14,15);
      // hoist all 4 B fragments (batched by the sched groups below)
      v16bf Bm[4];
#pragma unroll
      for (int t = 0; t < 4; ++t) {
        const unsigned short* bp =
            &in_lds[(ky * 66 + t * 16 + lm + kx) * WSTR + h * 8];
        v8bf blo = __builtin_bit_cast(v8bf, *(const u16x8*)&bp[0]);
        v8bf bhi = __builtin_bit_cast(v8bf, *(const u16x8*)&bp[16]);
        Bm[t] = __builtin_shufflevector(blo, bhi,
            0,1,2,3,4,5,6,7,8,9,10,11,12,13,14,15);
      }
#if !__has_builtin(__builtin_amdgcn_sched_group_barrier) && \
     __has_builtin(__builtin_amdgcn_sched_barrier)
      __builtin_amdgcn_sched_barrier(0);     // hard fence: loads above, WMMA below
#endif
      // 4 WMMAs share A: set reuse-A operand-cache hint on the last three.
      acc[0] = __builtin_amdgcn_wmma_f32_16x16x32_bf16(
          false, A, false, Bm[0], (short)0, acc[0], false, false);
      acc[1] = __builtin_amdgcn_wmma_f32_16x16x32_bf16(
          false, A, false, Bm[1], (short)0, acc[1], true, false);
      acc[2] = __builtin_amdgcn_wmma_f32_16x16x32_bf16(
          false, A, false, Bm[2], (short)0, acc[2], true, false);
      acc[3] = __builtin_amdgcn_wmma_f32_16x16x32_bf16(
          false, A, false, Bm[3], (short)0, acc[3], true, false);
#if __has_builtin(__builtin_amdgcn_sched_group_barrier)
      // pipeline shape per kk: burst of 10 DS reads, then 4 back-to-back WMMAs
      __builtin_amdgcn_sched_group_barrier(0x100, 10, 0);  // DS read group
      __builtin_amdgcn_sched_group_barrier(0x008, 4, 0);   // MFMA/WMMA group
#endif
    }
  }

  // fused epilogue: + noise_weight*noise + bias, leaky_relu(0.2) * sqrt(2)
  const float nw = *nwp;
  const float* nrow = noise + (size_t)b * (HH * WW) + (size_t)y0 * WW;
#pragma unroll
  for (int t = 0; t < 4; ++t) {
    int x = t * 16 + lm;
    float nv = nw * nrow[x];
#pragma unroll
    for (int r = 0; r < 8; ++r) {
      int m = r + h * 8;                         // C layout: M = r + 8*(lane/16)
      int cout = coutB + wv * 16 + m;
      float v = acc[t][r] + nv + bias[cout];
      v = (v < 0.f ? 0.2f * v : v) * 1.4142135623730951f;
      out[((size_t)(b * 512 + cout) * HH + y0) * WW + x] = v;
    }
  }
}

extern "C" void kernel_launch(void* const* d_in, const int* in_sizes, int n_in,
                              void* d_out, int out_size, void* d_ws, size_t ws_size,
                              hipStream_t stream) {
  (void)in_sizes; (void)n_in; (void)out_size; (void)ws_size;
  const float* content  = (const float*)d_in[0];
  const float* latent   = (const float*)d_in[1];
  const float* noise    = (const float*)d_in[2];
  const float* affine_w = (const float*)d_in[3];
  const float* affine_b = (const float*)d_in[4];
  const float* conv_w   = (const float*)d_in[5];
  const float* bias     = (const float*)d_in[6];
  const float* nwp      = (const float*)d_in[7];
  float* out = (float*)d_out;

  char* ws = (char*)d_ws;
  float* style          = (float*)ws;                                   // 32 KB
  float* dnorm          = (float*)(ws + 32768);                         // 32 KB
  unsigned short* wmod  = (unsigned short*)(ws + 65536);                // 75,497,472 B
  unsigned short* cbf   = (unsigned short*)(ws + 65536 + 75497472ull);  // 67,108,864 B

  k_style<<<32, 256, 0, stream>>>(latent, affine_w, affine_b, style);
  k_dnorm<<<1024, 256, 0, stream>>>(conv_w, style, dnorm);
  k_wmod<<<16384, 256, 0, stream>>>(conv_w, style, dnorm, wmod);
  k_cvt<<<dim3(4, 64, 16), 256, 0, stream>>>(content, cbf);
  k_conv<<<dim3(64, 4, 16), 256, 0, stream>>>(cbf, wmod, noise, bias, nwp, out);
}
